// VisionZipCompressor_67577015435650
// MI455X (gfx1250) — matrix operand order repaired and checked by reading, not compiled, not run on hardware
//
#include <hip/hip_runtime.h>
#include <hip/hip_bf16.h>
#include <cstdint>
#include <cstddef>

typedef _Float16 half_t;
typedef __attribute__((ext_vector_type(16))) _Float16 v16h;
typedef __attribute__((ext_vector_type(8)))  _Float16 v8h;
typedef __attribute__((ext_vector_type(8)))  float    v8f;
typedef int v4i __attribute__((vector_size(16)));
typedef __attribute__((address_space(1))) v4i* as1_v4i;
typedef __attribute__((address_space(3))) v4i* as3_v4i;

#define B_    16
#define H_    16
#define L_    577
#define NTOK  576
#define C_    1024
#define CTX   16
#define EPSF  1e-12f
#define INV_TAU_SIM  10.0f   // 1/0.1
#define INV_TAU_FEAT 5.0f    // 1/0.2
#define A_STRIDE 1032        // halves per LDS A-row (516 dwords == 4 mod 64 banks -> conflict-free)
#define SIM_LDS_BYTES (32 * A_STRIDE * 2)

// ---------------- block reductions (256 threads) ----------------
__device__ inline float blk_sum(float v, float* sh) {
    int tid = threadIdx.x;
    sh[tid] = v; __syncthreads();
    for (int s = 128; s > 0; s >>= 1) { if (tid < s) sh[tid] += sh[tid + s]; __syncthreads(); }
    float r = sh[0]; __syncthreads();
    return r;
}
__device__ inline float blk_max(float v, float* sh) {
    int tid = threadIdx.x;
    sh[tid] = v; __syncthreads();
    for (int s = 128; s > 0; s >>= 1) { if (tid < s) sh[tid] = fmaxf(sh[tid], sh[tid + s]); __syncthreads(); }
    float r = sh[0]; __syncthreads();
    return r;
}
__device__ inline float blk_min(float v, float* sh) {
    int tid = threadIdx.x;
    sh[tid] = v; __syncthreads();
    for (int s = 128; s > 0; s >>= 1) { if (tid < s) sh[tid] = fminf(sh[tid], sh[tid + s]); __syncthreads(); }
    float r = sh[0]; __syncthreads();
    return r;
}

// ---------------- kernel 1: CLS->patch attention, head mean ----------------
__global__ __launch_bounds__(576)
void vz_sattn_kernel(const float* __restrict__ attn, float* __restrict__ sattn) {
    const int b = blockIdx.x;
    const int j = threadIdx.x;            // 0..575
    float s = 0.f;
    #pragma unroll
    for (int h = 0; h < H_; ++h) {
        size_t idx = ((size_t)(b * H_ + h) * L_) * L_ + 1 + j; // row 0 (CLS), col 1+j
        s += attn[idx];
    }
    sattn[b * NTOK + j] = s * (1.0f / (float)H_);
}

// ---------------- kernel 2: L2-normalize keys -> z(f16), channel entropy ----------------
__global__ __launch_bounds__(256)
void vz_norment_kernel(const float* __restrict__ keys,
                       half_t* __restrict__ z16,
                       float* __restrict__ Hout) {
    __shared__ float red[256];
    const int b = blockIdx.x / NTOK;
    const int j = blockIdx.x % NTOK;
    const int tid = threadIdx.x;
    const float* row = keys + ((size_t)b * L_ + 1 + j) * C_;

    float x[4];
    float ss = 0.f;
    #pragma unroll
    for (int q = 0; q < 4; ++q) { x[q] = row[tid + q * 256]; ss += x[q] * x[q]; }
    ss = blk_sum(ss, red);
    const float inv = 1.0f / fmaxf(sqrtf(ss), EPSF);

    float z[4]; float mx = -3.0e38f;
    #pragma unroll
    for (int q = 0; q < 4; ++q) { z[q] = x[q] * inv; mx = fmaxf(mx, z[q]); }
    mx = blk_max(mx, red);

    float S = 0.f, T = 0.f;
    #pragma unroll
    for (int q = 0; q < 4; ++q) {
        float d = (z[q] - mx) * INV_TAU_FEAT;
        float e = __expf(d);
        S += e; T += (z[q] - mx) * e;
    }
    S = blk_sum(S, red);
    T = blk_sum(T, red);

    if (tid == 0) {
        float plogp = (T / S) * INV_TAU_FEAT - logf(S);       // sum p*log p
        Hout[b * NTOK + j] = -plogp / logf((float)C_);
    }
    half_t* zrow = z16 + ((size_t)b * NTOK + j) * C_;
    #pragma unroll
    for (int q = 0; q < 4; ++q) zrow[tid + q * 256] = (half_t)z[q];
}

// ---------------- online softmax-entropy helpers (tau = 0.1), branchless ----------------
__device__ inline void ent_update(float& m, float& S, float& T, float s) {
    float nm = fmaxf(m, s);
    float eo = __expf((m - nm) * INV_TAU_SIM);   // rescale old accumulators
    float en = __expf((s - nm) * INV_TAU_SIM);   // new element weight
    S = S * eo + en;
    T = T * eo + s * en;
    m = nm;
}
__device__ inline void ent_merge(float& m, float& S, float& T, float m2, float S2, float T2) {
    float M  = fmaxf(m, m2);
    float c1 = __expf((m  - M) * INV_TAU_SIM);
    float c2 = __expf((m2 - M) * INV_TAU_SIM);
    S = S * c1 + S2 * c2;
    T = T * c1 + T2 * c2;
    m = M;
}

// ---------------- kernel 3: sim = z z^T via WMMA + streaming row entropy ----------------
// grid (B, NTOK/32); block 256 = 8 waves; wave w: mi = w&1 (row 16-tile), ni = w>>1 (col 16-tile of 64-col step)
// A tile (32 rows x 1024 f16) staged in LDS via CDNA5 async global->LDS loads; B streamed from global.
__global__ __launch_bounds__(256)
void vz_sim_entropy_kernel(const half_t* __restrict__ z16, float* __restrict__ Iout) {
    extern __shared__ __align__(16) char smem[];     // SIM_LDS_BYTES
    half_t* Ash = (half_t*)smem;

    const int b     = blockIdx.x;
    const int rbase = blockIdx.y * 32;
    const int tid   = threadIdx.x;
    const int wave  = tid >> 5;
    const int lane  = tid & 31;
    const int l16   = lane & 15;
    const int hi    = lane >> 4;
    const int mi    = wave & 1;
    const int ni    = wave >> 1;
    const half_t* zb = z16 + (size_t)b * NTOK * C_;

    // async-stage A tile: 32 rows x 1024 halves, 16B chunks, padded LDS rows
    #pragma unroll
    for (int i = 0; i < 16; ++i) {
        const int id  = tid + i * 256;       // 0..4095
        const int row = id >> 7;             // 128 chunks per row
        const int cc  = id & 127;
        const half_t* g = zb + (size_t)(rbase + row) * C_ + cc * 8;
        half_t* l = Ash + row * A_STRIDE + cc * 8;
        __builtin_amdgcn_global_load_async_to_lds_b128((as1_v4i)g, (as3_v4i)l, 0, 0);
    }
    __builtin_amdgcn_s_wait_asynccnt(0);
    __syncthreads();

    float m[8], S[8], T[8];
    #pragma unroll
    for (int r = 0; r < 8; ++r) { m[r] = -3.0e38f; S[r] = 0.f; T[r] = 0.f; }

    const half_t* aL = Ash + (mi * 16 + l16) * A_STRIDE + hi * 16;  // A 16x32 f16 lane layout

    for (int nb = 0; nb < NTOK; nb += 64) {
        const int col = nb + ni * 16 + l16;
        const half_t* bptr = zb + (size_t)col * C_ + hi * 16;       // B 32x16 f16 lane layout
        v8f acc = {};
        #pragma unroll 4
        for (int kk = 0; kk < C_; kk += 32) {
            union { v16h v; v8h h2[2]; } A, Bm;
            A.h2[0] = *(const v8h*)(aL + kk);
            A.h2[1] = *(const v8h*)(aL + kk + 8);
            Bm.v    = *(const v16h*)(bptr + kk);
            acc = __builtin_amdgcn_wmma_f32_16x16x32_f16(
                    false, A.v, false, Bm.v, (short)0, acc, false, false);
        }
        const int gcol = nb + ni * 16 + l16;
        #pragma unroll
        for (int r = 0; r < 8; ++r) {
            const int grow = rbase + mi * 16 + hi * 8 + r;   // D layout: VGPR r -> M = r + hi*8
            float s = acc[r];
            if (gcol == grow) s = -1e9f;                     // diag mask (cndmask)
            ent_update(m[r], S[r], T[r], s);
        }
    }
    __syncthreads();   // done reading Ash; overlay reduction arrays on same LDS

    float (*Lm)[64] = (float(*)[64])(smem);
    float (*LS)[64] = (float(*)[64])(smem + 8192);
    float (*LT)[64] = (float(*)[64])(smem + 16384);

    // partials: row (0..31) x source (0..63)
    #pragma unroll
    for (int r = 0; r < 8; ++r) {
        const int rr = mi * 16 + hi * 8 + r;
        const int sc = ni * 16 + l16;
        Lm[rr][sc] = m[r]; LS[rr][sc] = S[r]; LT[rr][sc] = T[r];
    }
    __syncthreads();

    // stage 1: 8 partials per thread
    {
        const int row = tid >> 3, part = tid & 7;
        float mm = Lm[row][part * 8], ss = LS[row][part * 8], tt = LT[row][part * 8];
        #pragma unroll
        for (int q = 1; q < 8; ++q)
            ent_merge(mm, ss, tt, Lm[row][part * 8 + q], LS[row][part * 8 + q], LT[row][part * 8 + q]);
        __syncthreads();
        Lm[row][part] = mm; LS[row][part] = ss; LT[row][part] = tt;
    }
    __syncthreads();

    // stage 2: final 8 -> entropy -> I score
    if (tid < 32) {
        float mm = Lm[tid][0], ss = LS[tid][0], tt = LT[tid][0];
        #pragma unroll
        for (int q = 1; q < 8; ++q)
            ent_merge(mm, ss, tt, Lm[tid][q], LS[tid][q], LT[tid][q]);
        float qlogq = (tt / ss - mm) * INV_TAU_SIM - logf(ss); // sum q*log q
        float Hsim  = -qlogq / logf((float)NTOK);
        Iout[b * NTOK + rbase + tid] = 1.0f - Hsim;
    }
}

// ---------------- kernel 4: combine scores, top-k membership, keep/remain plan ----------------
__global__ __launch_bounds__(256)
void vz_plan_kernel(const float* __restrict__ sattn, const float* __restrict__ Hent,
                    const float* __restrict__ Isc,
                    int* __restrict__ keep_pos, int* __restrict__ remain_pos, int k) {
    __shared__ float score[NTOK];
    __shared__ int   kept[L_];
    __shared__ float red[256];

    const int b = blockIdx.x, tid = threadIdx.x;
    const float* sa = sattn + b * NTOK;
    const float* he = Hent  + b * NTOK;
    const float* ii = Isc   + b * NTOK;

    float mn0 = 3e38f, mx0 = -3e38f, mn1 = 3e38f, mx1 = -3e38f, mn2 = 3e38f, mx2 = -3e38f;
    for (int j = tid; j < NTOK; j += 256) {
        float a = sa[j], h = he[j], c = ii[j];
        mn0 = fminf(mn0, a); mx0 = fmaxf(mx0, a);
        mn1 = fminf(mn1, h); mx1 = fmaxf(mx1, h);
        mn2 = fminf(mn2, c); mx2 = fmaxf(mx2, c);
    }
    mn0 = blk_min(mn0, red); mx0 = blk_max(mx0, red);
    mn1 = blk_min(mn1, red); mx1 = blk_max(mx1, red);
    mn2 = blk_min(mn2, red); mx2 = blk_max(mx2, red);
    const float r0 = 1.0f / (mx0 - mn0 + EPSF);
    const float r1 = 1.0f / (mx1 - mn1 + EPSF);
    const float r2 = 1.0f / (mx2 - mn2 + EPSF);

    for (int j = tid; j < NTOK; j += 256)
        score[j] = (sa[j] - mn0) * r0 + 0.4f * ((he[j] - mn1) * r1) + 0.6f * ((ii[j] - mn2) * r2);
    __syncthreads();

    // rank-based top-k (ties -> lower index wins, matching lax.top_k)
    for (int j = tid; j < NTOK; j += 256) {
        float v = score[j];
        int rank = 0;
        for (int q = 0; q < NTOK; ++q) {
            float u = score[q];
            rank += (u > v) || (u == v && q < j);
        }
        kept[1 + j] = (rank < k) ? 1 : 0;
    }
    if (tid == 0) kept[0] = 1;  // CLS
    __syncthreads();

    for (int p = tid; p < L_; p += 256) {
        int cnt = 0;
        for (int q = 0; q < p; ++q) cnt += kept[q];
        if (kept[p]) keep_pos[b * 80 + cnt] = p;
        else         remain_pos[b * NTOK + (p - cnt)] = p;
    }
}

// ---------------- kernel 5: gather dominant tokens ----------------
__global__ __launch_bounds__(256)
void vz_dominant_kernel(const float* __restrict__ hidden, const int* __restrict__ keep_pos,
                        float* __restrict__ out, int k) {
    const int b = blockIdx.x, i = blockIdx.y, tid = threadIdx.x;
    const int p = keep_pos[b * 80 + i];
    const float* src = hidden + ((size_t)b * L_ + p) * C_;
    float* dst = out + ((size_t)b * (k + 1 + CTX) + i) * C_;
    #pragma unroll
    for (int q = 0; q < 4; ++q) dst[tid + q * 256] = src[tid + q * 256];
}

// ---------------- kernel 6: assign merge tokens to nearest target ----------------
__global__ __launch_bounds__(256)
void vz_assign_kernel(const half_t* __restrict__ z16, const int* __restrict__ remain_pos,
                      int* __restrict__ assign, int n_remain, int step) {
    __shared__ half_t tg[CTX * C_];   // 32KB target rows
    const int b = blockIdx.x, tid = threadIdx.x;

    for (int t = 0; t < CTX; ++t) {
        const int rp = remain_pos[b * NTOK + t * step];       // position in [1,L)
        const half_t* src = z16 + ((size_t)b * NTOK + (rp - 1)) * C_;
        for (int c = tid; c < C_; c += 256) tg[t * C_ + c] = src[c];
    }
    __syncthreads();

    const int wave = tid >> 5, lane = tid & 31;
    const int i = blockIdx.y * 8 + wave;
    if (i >= n_remain) return;
    const bool is_tgt = (i < CTX * step) && (i % step == 0);
    if (is_tgt) { if (lane == 0) assign[b * NTOK + i] = -1; return; }

    const int rp = remain_pos[b * NTOK + i];
    const half_t* tok = z16 + ((size_t)b * NTOK + (rp - 1)) * C_;

    float acc[CTX];
    #pragma unroll
    for (int t = 0; t < CTX; ++t) acc[t] = 0.f;
    for (int cc = 0; cc < C_ / 32; ++cc) {
        const int c = lane + cc * 32;
        const float tv = (float)tok[c];
        #pragma unroll
        for (int t = 0; t < CTX; ++t) acc[t] += tv * (float)tg[t * C_ + c];
    }
    #pragma unroll
    for (int t = 0; t < CTX; ++t) {
        float v = acc[t];
        for (int off = 16; off > 0; off >>= 1) v += __shfl_xor(v, off, 32);
        acc[t] = v;
    }
    if (lane == 0) {
        int best = 0; float bv = acc[0];
        #pragma unroll
        for (int t = 1; t < CTX; ++t) if (acc[t] > bv) { bv = acc[t]; best = t; }
        assign[b * NTOK + i] = best;
    }
}

// ---------------- kernel 7: segment mean -> ctx tokens ----------------
__global__ __launch_bounds__(256)
void vz_ctx_kernel(const float* __restrict__ hidden, const int* __restrict__ remain_pos,
                   const int* __restrict__ assign, float* __restrict__ out,
                   int n_remain, int step, int k) {
    const int b = blockIdx.x, a = blockIdx.y, tid = threadIdx.x;
    float sum[4] = {0.f, 0.f, 0.f, 0.f};
    int cnt = 0;
    for (int i = 0; i < n_remain; ++i) {
        if (assign[b * NTOK + i] == a) {
            ++cnt;
            const int p = remain_pos[b * NTOK + i];
            const float* src = hidden + ((size_t)b * L_ + p) * C_;
            #pragma unroll
            for (int q = 0; q < 4; ++q) sum[q] += src[tid + q * 256];
        }
    }
    float* dst = out + ((size_t)b * (k + 1 + CTX) + (k + 1 + a)) * C_;
    if (cnt > 0) {
        const float inv = 1.0f / (float)cnt;
        #pragma unroll
        for (int q = 0; q < 4; ++q) dst[tid + q * 256] = sum[q] * inv;
    } else {
        const int p = remain_pos[b * NTOK + a * step];
        const float* src = hidden + ((size_t)b * L_ + p) * C_;
        #pragma unroll
        for (int q = 0; q < 4; ++q) dst[tid + q * 256] = src[tid + q * 256];
    }
}

// ---------------- host launcher ----------------
extern "C" void kernel_launch(void* const* d_in, const int* in_sizes, int n_in,
                              void* d_out, int out_size, void* d_ws, size_t ws_size,
                              hipStream_t stream) {
    const float* attn   = (const float*)d_in[0];
    const float* keys   = (const float*)d_in[1];
    const float* hidden = (const float*)d_in[2];
    float* out = (float*)d_out;
    (void)in_sizes; (void)n_in; (void)ws_size;

    // k derived from output shape: out_size = B * (k + 1 + CTX) * C
    int k = out_size / (B_ * C_) - (CTX + 1);
    if (k < 4) k = 4; if (k > 64) k = 64;
    const int n_remain = NTOK - k;
    int step = n_remain / CTX; if (step < 1) step = 1;

    // workspace layout
    char* ws = (char*)d_ws;
    const size_t zBytes = (size_t)B_ * NTOK * C_ * sizeof(half_t);     // ~18.9 MB
    half_t* z16      = (half_t*)(ws);
    float*  sattn    = (float*)(ws + zBytes);
    float*  Hent     = sattn + B_ * NTOK;
    float*  Iscore   = Hent  + B_ * NTOK;
    int*    keep_pos = (int*)(Iscore + B_ * NTOK);                     // [B][80]
    int*    remainp  = keep_pos + B_ * 80;                             // [B][576]
    int*    assignp  = remainp  + B_ * NTOK;                           // [B][576]

    vz_sattn_kernel<<<dim3(B_), dim3(576), 0, stream>>>(attn, sattn);
    vz_norment_kernel<<<dim3(B_ * NTOK), dim3(256), 0, stream>>>(keys, z16, Hent);
    vz_sim_entropy_kernel<<<dim3(B_, NTOK / 32), dim3(256), SIM_LDS_BYTES, stream>>>(z16, Iscore);
    vz_plan_kernel<<<dim3(B_), dim3(256), 0, stream>>>(sattn, Hent, Iscore, keep_pos, remainp, k);
    vz_dominant_kernel<<<dim3(B_, k + 1), dim3(256), 0, stream>>>(hidden, keep_pos, out, k);
    vz_assign_kernel<<<dim3(B_, (n_remain + 7) / 8), dim3(256), 0, stream>>>(z16, remainp, assignp, n_remain, step);
    vz_ctx_kernel<<<dim3(B_, CTX), dim3(256), 0, stream>>>(hidden, remainp, assignp, out, n_remain, step, k);
}
